// GraphEncoder_2061584302388
// MI455X (gfx1250) — matrix operand order repaired and checked by reading, not compiled, not run on hardware
//
#include <hip/hip_runtime.h>

typedef __attribute__((ext_vector_type(16))) __bf16 bf16x16;
typedef __attribute__((ext_vector_type(8)))  float  f32x8;
typedef __attribute__((ext_vector_type(8)))  unsigned int u32x8;
typedef __attribute__((ext_vector_type(4)))  unsigned int u32x4;

#define HID 64
#define DIN 128

union FragU { u32x4 q[2]; u32x8 u; bf16x16 h; };

__device__ __forceinline__ unsigned pack2bf(float a, float b) {
  union { __bf16 h[2]; unsigned u; } t;
  t.h[0] = (__bf16)a;
  t.h[1] = (__bf16)b;
  return t.u;
}

__device__ __forceinline__ f32x8 wmma_bf16(bf16x16 a, bf16x16 b, f32x8 c) {
  return __builtin_amdgcn_wmma_f32_16x16x32_bf16(false, a, false, b, (short)0, c, false, false);
}

// A fragment: 16x32 bf16 from an LDS tile stored as packed K-pair words
// (row length ldW words). Per ISA 7.12.2 the lane's 8 words are two runs
// of 4: m*ldW + 16*kt + 4*grp + [0..3] and +8  ->  two b128 loads.
__device__ __forceinline__ bf16x16 load_A_frag(const unsigned* t32, int ktile, int ldW) {
  int lane = threadIdx.x & 31;
  int m = lane & 15;
  int grp = lane >> 4;
  const unsigned* base = t32 + m * ldW + ktile * 16 + 4 * grp;
  FragU r;
  r.q[0] = *(const u32x4*)(base);
  r.q[1] = *(const u32x4*)(base + 8);
  return r.h;
}

// B fragment from fragment-major packed weights in GLOBAL memory:
// word idx = (((kt*4 + ntile)*32 + lane)*8 + v)  ->  two global b128 loads.
__device__ __forceinline__ bf16x16 load_B_frag(const unsigned* __restrict__ Wp,
                                               int ktile, int ntile) {
  int lane = threadIdx.x & 31;
  const u32x4* p = (const u32x4*)(Wp + ((((ktile * 4 + ntile) * 32 + lane)) << 3));
  FragU r;
  r.q[0] = p[0];
  r.q[1] = p[1];
  return r.h;
}

__device__ __forceinline__ void atomicMaxFloat(float* addr, float v) {
  if (v >= 0.0f) atomicMax((int*)addr, __float_as_int(v));
  else           atomicMin((unsigned int*)addr, (unsigned int)__float_as_int(v));
}

// ---------------------------------------------------------------- pack
// Repack W[K][64] (f32 row-major) into fragment-major packed bf16 pairs.
__global__ void k_pack(const float* __restrict__ W, unsigned* __restrict__ out,
                       int Kt) {
  int i = blockIdx.x * blockDim.x + threadIdx.x;
  if (i >= Kt * 1024) return;
  int v = i & 7;
  int lane = (i >> 3) & 31;
  int ntile = (i >> 8) & 3;
  int kt = i >> 10;
  int p = kt * 16 + (v >> 2) * 8 + (lane >> 4) * 4 + (v & 3);
  int n = ntile * 16 + (lane & 15);
  out[i] = pack2bf(W[(2 * p) * HID + n], W[(2 * p + 1) * HID + n]);
}

// ---------------------------------------------------------------- kernel 1
// h = relu(LN(x @ Win + bin)); also writes bf16 mirror hb
__global__ __launch_bounds__(256) void k_inproj(
    const float* __restrict__ x, const unsigned* __restrict__ Wp,
    const float* __restrict__ bias, const float* __restrict__ g,
    const float* __restrict__ beta, float* __restrict__ h,
    __bf16* __restrict__ hb, int N) {
  __shared__ __align__(16) unsigned sA[8][16 * DIN / 2];  // 32 KB

  int tid = threadIdx.x;
  int wave = tid >> 5, lane = tid & 31;
  int row0 = blockIdx.x * 128 + wave * 16;
  unsigned* tile = sA[wave];
  for (int i = lane; i < 16 * DIN / 2; i += 32) {
    int r = i >> 6, c2 = i & 63;
    int rr = min(row0 + r, N - 1);
    const float2* xp = (const float2*)(x + (long)rr * DIN);
    float2 xx = xp[c2];
    tile[i] = pack2bf(xx.x, xx.y);
  }
  bf16x16 a0 = load_A_frag(tile, 0, DIN / 2);
  bf16x16 a1 = load_A_frag(tile, 1, DIN / 2);
  bf16x16 a2 = load_A_frag(tile, 2, DIN / 2);
  bf16x16 a3 = load_A_frag(tile, 3, DIN / 2);

  f32x8 acc[4];
  int n = lane & 15, grp = lane >> 4;
#pragma unroll
  for (int nt = 0; nt < 4; ++nt) {
    f32x8 c = {};
    c = wmma_bf16(a0, load_B_frag(Wp, 0, nt), c);
    c = wmma_bf16(a1, load_B_frag(Wp, 1, nt), c);
    c = wmma_bf16(a2, load_B_frag(Wp, 2, nt), c);
    c = wmma_bf16(a3, load_B_frag(Wp, 3, nt), c);
#pragma unroll
    for (int r = 0; r < 8; ++r) acc[nt][r] = c[r] + bias[nt * 16 + n];
  }
#pragma unroll
  for (int r = 0; r < 8; ++r) {
    float s  = acc[0][r] + acc[1][r] + acc[2][r] + acc[3][r];
    float s2 = acc[0][r] * acc[0][r] + acc[1][r] * acc[1][r] +
               acc[2][r] * acc[2][r] + acc[3][r] * acc[3][r];
#pragma unroll
    for (int off = 1; off < 16; off <<= 1) {
      s += __shfl_xor(s, off);
      s2 += __shfl_xor(s2, off);
    }
    float mean = s * (1.0f / 64.0f);
    float var  = s2 * (1.0f / 64.0f) - mean * mean;
    float inv  = rsqrtf(var + 1e-5f);
    int row = row0 + r + 8 * grp;
    if (row < N) {
#pragma unroll
      for (int nt = 0; nt < 4; ++nt) {
        int col = nt * 16 + n;
        float y = (acc[nt][r] - mean) * inv * g[col] + beta[col];
        y = fmaxf(y, 0.0f);
        h[(long)row * HID + col] = y;
        hb[(long)row * HID + col] = (__bf16)y;
      }
    }
  }
}

// ---------------------------------------------------------------- kernel 2
// q,k,v,skip = h @ W* + b*  (A tile async-copied from bf16 mirror;
// packed weights streamed straight from global, L2-resident)
__global__ __launch_bounds__(256) void k_qkvs(
    const __bf16* __restrict__ hb, const unsigned* __restrict__ wpk,
    const float* __restrict__ bq, const float* __restrict__ bk,
    const float* __restrict__ bv, const float* __restrict__ bsk,
    float* __restrict__ oq, float* __restrict__ ok,
    float* __restrict__ ov, float* __restrict__ os, int N) {
  __shared__ __align__(16) unsigned sA[8][16 * HID / 2];  // 16 KB

  int tid = threadIdx.x;
  int wave = tid >> 5, lane = tid & 31;
  int row0 = blockIdx.x * 128 + wave * 16;

  // async copy of this wave's 16x64 bf16 tile (2048 contiguous bytes);
  // low 32 bits of generic LDS address == wave-relative LDS offset
  unsigned ldsBase = (unsigned)(size_t)(&sA[wave][0]);
  long tileByte = (long)row0 * (HID * 2);
  long maxByte = (long)N * (HID * 2) - 16;
#pragma unroll
  for (int j = 0; j < 4; ++j) {
    long off = (long)j * 512 + (long)lane * 16;
    long gb = tileByte + off;
    if (gb > maxByte) gb = maxByte;  // clamp tail (rows >= N never stored)
    unsigned ldsAddr = ldsBase + (unsigned)off;
    const char* gp = (const char*)hb + gb;
    asm volatile("global_load_async_to_lds_b128 %0, %1, off"
                 :: "v"(ldsAddr), "v"(gp) : "memory");
  }
  asm volatile("s_wait_asynccnt 0" ::: "memory");

  unsigned* tile = sA[wave];
  bf16x16 a0 = load_A_frag(tile, 0, HID / 2);
  bf16x16 a1 = load_A_frag(tile, 1, HID / 2);

  float* outs[4] = {oq, ok, ov, os};
  const float* bsrc[4] = {bq, bk, bv, bsk};
  int n = lane & 15, grp = lane >> 4;
#pragma unroll
  for (int m = 0; m < 4; ++m) {
    const unsigned* Wp = wpk + m * 2048;  // 2 ktiles * 4 ntiles * 256 words
#pragma unroll
    for (int nt = 0; nt < 4; ++nt) {
      f32x8 c = {};
      c = wmma_bf16(a0, load_B_frag(Wp, 0, nt), c);
      c = wmma_bf16(a1, load_B_frag(Wp, 1, nt), c);
      float bval = bsrc[m][nt * 16 + n];
#pragma unroll
      for (int r = 0; r < 8; ++r) {
        int row = row0 + r + 8 * grp;
        if (row < N)
          outs[m][(long)row * HID + nt * 16 + n] = c[r] + bval;
      }
    }
  }
}

// ---------------------------------------------------------------- kernel 3
__global__ __launch_bounds__(256) void k_edge_logits(
    const int* __restrict__ srcI, const int* __restrict__ dstI,
    const float* __restrict__ q, const float* __restrict__ kk,
    float* __restrict__ logits, float* __restrict__ mbuf, int E) {
  int e = blockIdx.x * blockDim.x + threadIdx.x;
  if (e >= E) return;
  __builtin_prefetch(srcI + e + 2048, 0, 1);
  int s = srcI[e], d = dstI[e];
  const float4* qp = (const float4*)(q + (long)d * HID);
  const float4* kp = (const float4*)(kk + (long)s * HID);
  float acc[4] = {0.f, 0.f, 0.f, 0.f};
#pragma unroll
  for (int i = 0; i < 16; ++i) {
    float4 qa = qp[i], ka = kp[i];
    acc[i >> 2] += qa.x * ka.x + qa.y * ka.y + qa.z * ka.z + qa.w * ka.w;
  }
#pragma unroll
  for (int hh = 0; hh < 4; ++hh) {
    float lg = acc[hh] * 0.25f;  // 1/sqrt(16)
    logits[(long)e * 4 + hh] = lg;
    atomicMaxFloat(&mbuf[(long)d * 4 + hh], lg);
  }
}

// ---------------------------------------------------------------- kernel 4
__global__ __launch_bounds__(256) void k_edge_exp(
    const int* __restrict__ dstI, const float* __restrict__ mbuf,
    float* __restrict__ exbuf, float* __restrict__ denom, int E) {
  int e = blockIdx.x * blockDim.x + threadIdx.x;
  if (e >= E) return;
  int d = dstI[e];
#pragma unroll
  for (int hh = 0; hh < 4; ++hh) {
    float ex = __expf(exbuf[(long)e * 4 + hh] - mbuf[(long)d * 4 + hh]);
    exbuf[(long)e * 4 + hh] = ex;
    atomicAdd(&denom[(long)d * 4 + hh], ex);
  }
}

// ---------------------------------------------------------------- kernel 5
__global__ __launch_bounds__(256) void k_edge_agg(
    const int* __restrict__ srcI, const int* __restrict__ dstI,
    const float* __restrict__ exbuf, const float* __restrict__ denom,
    const float* __restrict__ vv, float* __restrict__ agg, int E) {
  long t = (long)blockIdx.x * blockDim.x + threadIdx.x;
  if (t >= (long)E * 4) return;
  long e = t >> 2;
  int hh = (int)(t & 3);
  int s = srcI[e], d = dstI[e];
  float alpha = exbuf[e * 4 + hh] / (denom[(long)d * 4 + hh] + 1e-16f);
  const float4* vp = (const float4*)(vv + (long)s * HID + hh * 16);
  float* ap = agg + (long)d * HID + hh * 16;
#pragma unroll
  for (int i = 0; i < 4; ++i) {
    float4 w = vp[i];
    atomicAdd(ap + i * 4 + 0, alpha * w.x);
    atomicAdd(ap + i * 4 + 1, alpha * w.y);
    atomicAdd(ap + i * 4 + 2, alpha * w.z);
    atomicAdd(ap + i * 4 + 3, alpha * w.w);
  }
}

// ---------------------------------------------------------------- kernel 6
__global__ __launch_bounds__(256) void k_node_update(
    float* __restrict__ h, __bf16* __restrict__ hb,
    const float* __restrict__ agg, const float* __restrict__ skip,
    const float* __restrict__ g, const float* __restrict__ beta, int N) {
  int node = (int)(((long)blockIdx.x * blockDim.x + threadIdx.x) >> 5);
  int lane = threadIdx.x & 31;
  if (node >= N) return;
  long base = (long)node * HID;
  float x0 = agg[base + lane] + skip[base + lane] + h[base + lane];
  float x1 = agg[base + 32 + lane] + skip[base + 32 + lane] + h[base + 32 + lane];
  float s = x0 + x1, s2 = x0 * x0 + x1 * x1;
#pragma unroll
  for (int off = 1; off < 32; off <<= 1) {
    s += __shfl_xor(s, off);
    s2 += __shfl_xor(s2, off);
  }
  float mean = s * (1.0f / 64.0f);
  float var = s2 * (1.0f / 64.0f) - mean * mean;
  float inv = rsqrtf(var + 1e-5f);
  float y0 = (x0 - mean) * inv * g[lane] + beta[lane];
  float y1 = (x1 - mean) * inv * g[32 + lane] + beta[32 + lane];
  y0 = y0 > 0.f ? y0 : __expf(y0) - 1.f;
  y1 = y1 > 0.f ? y1 : __expf(y1) - 1.f;
  h[base + lane] = y0;
  h[base + 32 + lane] = y1;
  hb[base + lane] = (__bf16)y0;
  hb[base + 32 + lane] = (__bf16)y1;
}

// ---------------------------------------------------------------- kernel 7
__global__ __launch_bounds__(256) void k_pool_scatter(
    const float* __restrict__ h, const int* __restrict__ batch,
    float* __restrict__ mx, float* __restrict__ sm, float* __restrict__ cnt,
    int N) {
  int node = blockIdx.x * blockDim.x + threadIdx.x;
  if (node >= N) return;
  int gi = batch[node];
  const float4* hp = (const float4*)(h + (long)node * HID);
  float4 vals[16];
  float s = 0.f;
#pragma unroll
  for (int i = 0; i < 16; ++i) {
    vals[i] = hp[i];
    s += vals[i].x + vals[i].y + vals[i].z + vals[i].w;
  }
  float score = 1.0f / (1.0f + __expf(-s));
  float* mxp = mx + gi * HID;
  float* smp = sm + gi * HID;
#pragma unroll
  for (int i = 0; i < 16; ++i) {
    float4 w = vals[i];
    float a0 = w.x * score, a1 = w.y * score, a2 = w.z * score, a3 = w.w * score;
    atomicMaxFloat(mxp + i * 4 + 0, a0); atomicAdd(smp + i * 4 + 0, a0);
    atomicMaxFloat(mxp + i * 4 + 1, a1); atomicAdd(smp + i * 4 + 1, a1);
    atomicMaxFloat(mxp + i * 4 + 2, a2); atomicAdd(smp + i * 4 + 2, a2);
    atomicMaxFloat(mxp + i * 4 + 3, a3); atomicAdd(smp + i * 4 + 3, a3);
  }
  atomicAdd(&cnt[gi], 1.0f);
}

// ---------------------------------------------------------------- kernel 8
__global__ __launch_bounds__(64) void k_final(
    const float* __restrict__ mx, const float* __restrict__ sm,
    const float* __restrict__ cnt, const float* __restrict__ W1,
    const float* __restrict__ b1, const float* __restrict__ W2,
    const float* __restrict__ b2, float* __restrict__ out) {
  int g = threadIdx.x;  // one graph row per thread
  __shared__ float pooled[64][2 * HID];
  __shared__ float o1[64][HID];
  float c = cnt[g];
  float invc = 1.0f / fmaxf(c, 1.0f);
  for (int i = 0; i < HID; ++i) {
    float m = mx[g * HID + i];
    pooled[g][i] = (c > 0.f) ? m : 0.f;
    pooled[g][HID + i] = sm[g * HID + i] * invc;
  }
  for (int j = 0; j < HID; ++j) {
    float a = b1[j];
    for (int i = 0; i < 2 * HID; ++i) a += pooled[g][i] * W1[i * HID + j];
    o1[g][j] = a > 0.f ? a : __expf(a) - 1.f;
  }
  float o[HID];
  float nrm = 0.f;
  for (int j = 0; j < HID; ++j) {
    float a = b2[j];
    for (int i = 0; i < HID; ++i) a += o1[g][i] * W2[i * HID + j];
    o[j] = a;
    nrm += a * a;
  }
  nrm = fmaxf(sqrtf(nrm), 1e-12f);
  for (int j = 0; j < HID; ++j) out[g * HID + j] = o[j] / nrm;
}

// ---------------------------------------------------------------- utility
__global__ void k_fill(float* __restrict__ p, float val, long n) {
  long i = (long)blockIdx.x * blockDim.x + threadIdx.x;
  if (i < n) p[i] = val;
}

// ---------------------------------------------------------------- launch
extern "C" void kernel_launch(void* const* d_in, const int* in_sizes, int n_in,
                              void* d_out, int out_size, void* d_ws,
                              size_t ws_size, hipStream_t stream) {
  const float* x     = (const float*)d_in[0];
  const int*   eidx  = (const int*)d_in[1];
  const int*   batch = (const int*)d_in[2];
  int N = in_sizes[2];
  int E = in_sizes[1] / 2;
  const int* srcI = eidx;
  const int* dstI = eidx + E;

  const float* inW  = (const float*)d_in[3];
  const float* inb  = (const float*)d_in[4];
  const float* ing  = (const float*)d_in[5];
  const float* inbt = (const float*)d_in[6];
  const float* W1 = (const float*)d_in[37];
  const float* b1 = (const float*)d_in[38];
  const float* W2 = (const float*)d_in[39];
  const float* b2 = (const float*)d_in[40];

  float* ws = (float*)d_ws;
  long off = 0;
  float* h     = ws + off; off += (long)N * HID;
  __bf16* hb   = (__bf16*)(ws + off); off += (long)N * HID / 2;
  float* hq    = ws + off; off += (long)N * HID;
  float* hk    = ws + off; off += (long)N * HID;
  float* hv    = ws + off; off += (long)N * HID;
  float* hs    = ws + off; off += (long)N * HID;
  float* agg   = ws + off; off += (long)N * HID;
  float* exbuf = ws + off; off += (long)E * 4;
  float* mbuf  = ws + off; off += (long)N * 4;
  float* denom = ws + off; off += (long)N * 4;
  float* mxb   = ws + off; off += 64 * HID;
  float* smb   = ws + off; off += 64 * HID;
  float* cntb  = ws + off; off += 64;
  unsigned* wpkIn = (unsigned*)(ws + off); off += 4096;      // 4 ktiles
  unsigned* wpkGat[3];
  for (int L = 0; L < 3; ++L) { wpkGat[L] = (unsigned*)(ws + off); off += 8192; }

  const float NEG_INF = -__builtin_huge_valf();
  int gridRows = (N + 127) / 128;
  int gridE = (E + 255) / 256;

  // repack all weights into fragment-major bf16-pair layout
  k_pack<<<(4096 + 255) / 256, 256, 0, stream>>>(inW, wpkIn, 4);
  for (int L = 0; L < 3; ++L) {
    int base = 7 + L * 8;
    for (int m = 0; m < 4; ++m) {
      const float* Wm = (const float*)d_in[base + 2 * m];
      k_pack<<<(2048 + 255) / 256, 256, 0, stream>>>(Wm, wpkGat[L] + m * 2048, 2);
    }
  }

  k_inproj<<<gridRows, 256, 0, stream>>>(x, wpkIn, inb, ing, inbt, h, hb, N);

  for (int L = 0; L < 3; ++L) {
    int base = 7 + L * 8;
    const float* bq  = (const float*)d_in[base + 1];
    const float* bk  = (const float*)d_in[base + 3];
    const float* bv  = (const float*)d_in[base + 5];
    const float* bsk = (const float*)d_in[base + 7];
    const float* lng = (const float*)d_in[31 + L * 2];
    const float* lnb = (const float*)d_in[31 + L * 2 + 1];

    long n4 = (long)N * 4, n64 = (long)N * HID;
    k_fill<<<(int)((n4 + 255) / 256), 256, 0, stream>>>(mbuf, NEG_INF, n4);
    k_fill<<<(int)((n4 + 255) / 256), 256, 0, stream>>>(denom, 0.f, n4);
    k_fill<<<(int)((n64 + 255) / 256), 256, 0, stream>>>(agg, 0.f, n64);

    k_qkvs<<<gridRows, 256, 0, stream>>>(hb, wpkGat[L], bq, bk, bv, bsk,
                                         hq, hk, hv, hs, N);
    k_edge_logits<<<gridE, 256, 0, stream>>>(srcI, dstI, hq, hk, exbuf, mbuf, E);
    k_edge_exp<<<gridE, 256, 0, stream>>>(dstI, mbuf, exbuf, denom, E);
    long tE = (long)E * 4;
    k_edge_agg<<<(int)((tE + 255) / 256), 256, 0, stream>>>(srcI, dstI, exbuf,
                                                            denom, hv, agg, E);
    k_node_update<<<(N + 7) / 8, 256, 0, stream>>>(h, hb, agg, hs, lng, lnb, N);
  }

  k_fill<<<(64 * HID + 255) / 256, 256, 0, stream>>>(mxb, NEG_INF, 64 * HID);
  k_fill<<<(64 * HID + 255) / 256, 256, 0, stream>>>(smb, 0.f, 64 * HID);
  k_fill<<<1, 64, 0, stream>>>(cntb, 0.f, 64);
  k_pool_scatter<<<(N + 255) / 256, 256, 0, stream>>>(h, batch, mxb, smb, cntb, N);
  k_final<<<1, 64, 0, stream>>>(mxb, smb, cntb, W1, b1, W2, b2, (float*)d_out);
}
